// PairBiasAttention_15564961481064
// MI455X (gfx1250) — matrix operand order repaired and checked by reading, not compiled, not run on hardware
//
#include <hip/hip_runtime.h>
#include <hip/hip_bf16.h>

typedef _Float16 half_t;
typedef __attribute__((ext_vector_type(16))) _Float16 v16h;
typedef __attribute__((ext_vector_type(8)))  _Float16 v8h;
typedef __attribute__((ext_vector_type(8)))  float    v8f;

__device__ __forceinline__ v8f wmma_f16(v16h a, v16h b, v8f c) {
    return __builtin_amdgcn_wmma_f32_16x16x32_f16(false, a, false, b, (short)0, c,
                                                  false, false);
}

// Build A-fragment (16x32 f16): halves j<16 at base[j] (contiguous).
__device__ __forceinline__ v16h load_a_frag(const half_t* base) {
    v8h lo = *(const v8h*)(base);
    v8h hi = *(const v8h*)(base + 8);
    return __builtin_shufflevector(lo, hi, 0, 1, 2, 3, 4, 5, 6, 7,
                                           8, 9, 10, 11, 12, 13, 14, 15);
}

// ---------------------------------------------------------------------------
// Prepass: f32 -> f16 elementwise (x, in_proj_w, out_w), 8 elems / thread.
// ---------------------------------------------------------------------------
__global__ __launch_bounds__(256) void cvt_kernel(const float* __restrict__ in,
                                                  half_t* __restrict__ out, int n) {
  const int i = (blockIdx.x * 256 + threadIdx.x) * 8;
  if (i >= n) return;
  const float4 a = *(const float4*)(in + i);
  const float4 b = *(const float4*)(in + i + 4);
  v8h o;
  o[0] = (half_t)a.x; o[1] = (half_t)a.y; o[2] = (half_t)a.z; o[3] = (half_t)a.w;
  o[4] = (half_t)b.x; o[5] = (half_t)b.y; o[6] = (half_t)b.z; o[7] = (half_t)b.w;
  *(v8h*)(out + i) = o;
}

// ---------------------------------------------------------------------------
// Kernel 1: fused QKV projection. qkv = x @ in_proj_w^T + in_proj_b
// M=4096, N=1536, K=512.  Block: 128(M) x 64(N), 4 waves; wave: 32(M) x 64(N).
// n0 is 64-aligned => the whole block hits exactly one of {Q,K,V} and one head.
// ---------------------------------------------------------------------------
__global__ __launch_bounds__(128) void qkv_kernel(
    const half_t* __restrict__ xh, const half_t* __restrict__ wh,
    const float* __restrict__ bias,
    half_t* __restrict__ Q, half_t* __restrict__ Kt, half_t* __restrict__ Vt) {
  const int lane = threadIdx.x & 31;
  const int wave = threadIdx.x >> 5;
  const int c  = lane & 15;
  const int hi = lane >> 4;
  const int m0 = blockIdx.x * 128 + wave * 32;
  const int n0 = blockIdx.y * 64;

  v8f acc[2][4] = {};
  for (int k0 = 0; k0 < 512; k0 += 32) {
    // unconditional speculative prefetch of next k-tile (safe past the end)
    __builtin_prefetch(xh + (size_t)(m0 + c) * 512 + k0 + 32, 0, 3);
    __builtin_prefetch(wh + (size_t)(n0 + c) * 512 + k0 + 32, 0, 3);
    v16h af[2];
#pragma unroll
    for (int r = 0; r < 2; ++r)
      af[r] = load_a_frag(xh + (size_t)(m0 + r * 16 + c) * 512 + k0 + hi * 16);
    v16h bf[4];
#pragma unroll
    for (int t = 0; t < 4; ++t)
      bf[t] = *(const v16h*)(wh + (size_t)(n0 + t * 16 + c) * 512 + k0 + hi * 16);
#pragma unroll
    for (int r = 0; r < 2; ++r)
#pragma unroll
      for (int t = 0; t < 4; ++t) acc[r][t] = wmma_f16(af[r], bf[t], acc[r][t]);
  }

  // epilogue: section / head are block-constants (n0 is 64-aligned)
  const int sec = n0 >> 9;              // 0=q 1=k 2=v
  const int h   = (n0 & 511) >> 6;      // head
  float bcol[4];
#pragma unroll
  for (int t = 0; t < 4; ++t) bcol[t] = bias[n0 + t * 16 + c];

  if (sec == 0) {
#pragma unroll
    for (int r = 0; r < 2; ++r)
#pragma unroll
      for (int t = 0; t < 4; ++t)
#pragma unroll
        for (int v = 0; v < 8; ++v) {
          const int m = m0 + r * 16 + hi * 8 + v;
          const int b = m >> 10, l = m & 1023;
          Q[(((size_t)(b * 8 + h)) * 1024 + l) * 64 + t * 16 + c] =
              (half_t)((acc[r][t][v] + bcol[t]) * 0.125f);
        }
  } else if (sec == 1) {
#pragma unroll
    for (int r = 0; r < 2; ++r)
#pragma unroll
      for (int t = 0; t < 4; ++t)
#pragma unroll
        for (int v = 0; v < 8; ++v) {
          const int m = m0 + r * 16 + hi * 8 + v;
          const int b = m >> 10, l = m & 1023;
          Kt[(((size_t)(b * 8 + h)) * 1024 + l) * 64 + t * 16 + c] =
              (half_t)(acc[r][t][v] + bcol[t]);
        }
  } else {
#pragma unroll
    for (int r = 0; r < 2; ++r)
#pragma unroll
      for (int t = 0; t < 4; ++t)
#pragma unroll
        for (int v = 0; v < 8; ++v) {
          const int m = m0 + r * 16 + hi * 8 + v;
          const int b = m >> 10, l = m & 1023;
          Vt[(((size_t)(b * 8 + h)) * 64 + t * 16 + c) * 1024 + l] =
              (half_t)(acc[r][t][v] + bcol[t]);
        }
  }
}

// ---------------------------------------------------------------------------
// Kernel 2: flash attention with pair bias.
// One block per (b*H+h, q-tile of 64); 4 waves, each wave owns 16 q-rows.
// ---------------------------------------------------------------------------
__global__ __launch_bounds__(128) void attn_kernel(
    const half_t* __restrict__ Q, const half_t* __restrict__ Kt,
    const half_t* __restrict__ Vt, const float* __restrict__ pair_bias,
    const float* __restrict__ pair_w, const float* __restrict__ pair_b,
    half_t* __restrict__ O2) {
  __shared__ half_t pbuf[4][16 * 32];   // per-wave P staging (D-layout -> A-layout)

  const int lane = threadIdx.x & 31;
  const int wave = threadIdx.x >> 5;
  const int c  = lane & 15;
  const int hi = lane >> 4;
  const int bh = blockIdx.x;
  const int b  = bh >> 3, h = bh & 7;
  const int q0 = blockIdx.y * 64 + wave * 16;

  const half_t* Qb = Q  + (size_t)bh * 1024 * 64;
  const half_t* Kb = Kt + (size_t)bh * 1024 * 64;
  const half_t* Vb = Vt + (size_t)bh * 64 * 1024;
  const float pw = pair_w[h];
  const float pb = pair_b[h];

  // Q A-fragments for both 32-wide k-steps of the hd=64 contraction (kept live)
  v16h qf[2];
#pragma unroll
  for (int s = 0; s < 2; ++s)
    qf[s] = load_a_frag(Qb + (size_t)(q0 + c) * 64 + s * 32 + hi * 16);

  v8f oacc[4] = {};
  float mrow[8], lrow[8];
#pragma unroll
  for (int v = 0; v < 8; ++v) { mrow[v] = -1e30f; lrow[v] = 0.f; }

  for (int j0 = 0; j0 < 1024; j0 += 32) {
    // prefetch next key/value chunk (speculative; safe past the end)
    __builtin_prefetch(Kb + (size_t)(j0 + 32 + c) * 64, 0, 3);
    __builtin_prefetch(Vb + (size_t)c * 1024 + j0 + 32, 0, 3);
    // ---- scores: init C with pair bias, then S = (Q/sqrt(hd)) K^T + C
    v8f s0, s1;
#pragma unroll
    for (int v = 0; v < 8; ++v) {
      const float* pr = pair_bias + ((size_t)b * 1024 + q0 + hi * 8 + v) * 1024 + j0;
      s0[v] = pr[c]      * pw + pb;
      s1[v] = pr[16 + c] * pw + pb;
    }
    // K^T B-fragments: 4 independent 32B vector loads, then 4 WMMAs
    v16h kf00 = *(const v16h*)(Kb + (size_t)(j0 + c)      * 64 +  0 + hi * 16);
    v16h kf01 = *(const v16h*)(Kb + (size_t)(j0 + c)      * 64 + 32 + hi * 16);
    v16h kf10 = *(const v16h*)(Kb + (size_t)(j0 + 16 + c) * 64 +  0 + hi * 16);
    v16h kf11 = *(const v16h*)(Kb + (size_t)(j0 + 16 + c) * 64 + 32 + hi * 16);
    s0 = wmma_f16(qf[0], kf00, s0);
    s0 = wmma_f16(qf[1], kf01, s0);
    s1 = wmma_f16(qf[0], kf10, s1);
    s1 = wmma_f16(qf[1], kf11, s1);
    // ---- online softmax update (per-row; 16-lane half-wave reductions)
#pragma unroll
    for (int v = 0; v < 8; ++v) {
      float mx = fmaxf(s0[v], s1[v]);
#pragma unroll
      for (int off = 8; off >= 1; off >>= 1) mx = fmaxf(mx, __shfl_xor(mx, off, 32));
      const float mnew  = fmaxf(mrow[v], mx);
      const float alpha = __expf(mrow[v] - mnew);
      mrow[v] = mnew;
      const float p0 = __expf(s0[v] - mnew);
      const float p1 = __expf(s1[v] - mnew);
      s0[v] = p0; s1[v] = p1;
      float rs = p0 + p1;
#pragma unroll
      for (int off = 8; off >= 1; off >>= 1) rs += __shfl_xor(rs, off, 32);
      lrow[v] = alpha * lrow[v] + rs;
#pragma unroll
      for (int t = 0; t < 4; ++t) oacc[t][v] *= alpha;
    }
    // ---- P: D-layout -> A-layout via per-wave LDS bounce (DS is wave-ordered)
    half_t* pl = pbuf[wave];
#pragma unroll
    for (int v = 0; v < 8; ++v) {
      const int r = hi * 8 + v;
      pl[r * 32 + c]      = (half_t)s0[v];
      pl[r * 32 + 16 + c] = (half_t)s1[v];
    }
    const v16h pfrag = load_a_frag(pl + c * 32 + hi * 16);
    // ---- O += P(16x32) @ V(32x64); Vt is [hd][L] so B-fragments are contiguous
    v16h vf[4];
#pragma unroll
    for (int t = 0; t < 4; ++t)
      vf[t] = *(const v16h*)(Vb + (size_t)(t * 16 + c) * 1024 + j0 + hi * 16);
#pragma unroll
    for (int t = 0; t < 4; ++t) oacc[t] = wmma_f16(pfrag, vf[t], oacc[t]);
  }
  // ---- finalize: one reciprocal per row, write O2[b][l][h*64+dd] fp16
#pragma unroll
  for (int v = 0; v < 8; ++v) {
    const float inv = 1.0f / lrow[v];
    const int l = q0 + hi * 8 + v;
#pragma unroll
    for (int t = 0; t < 4; ++t) {
      O2[((size_t)b * 1024 + l) * 512 + h * 64 + t * 16 + c] =
          (half_t)(oacc[t][v] * inv);
    }
  }
}

// ---------------------------------------------------------------------------
// Kernel 3: output projection.  out = O2 @ out_w^T + out_b  (fp32 result)
// M=4096, N=512, K=512.  Block: 128(M) x 64(N), wave: 32(M) x 64(N).
// ---------------------------------------------------------------------------
__global__ __launch_bounds__(128) void outproj_kernel(
    const half_t* __restrict__ O2, const half_t* __restrict__ wh,
    const float* __restrict__ bias, float* __restrict__ out) {
  const int lane = threadIdx.x & 31;
  const int wave = threadIdx.x >> 5;
  const int c  = lane & 15;
  const int hi = lane >> 4;
  const int m0 = blockIdx.x * 128 + wave * 32;
  const int n0 = blockIdx.y * 64;

  v8f acc[2][4] = {};
  for (int k0 = 0; k0 < 512; k0 += 32) {
    __builtin_prefetch(O2 + (size_t)(m0 + c) * 512 + k0 + 32, 0, 3);
    __builtin_prefetch(wh + (size_t)(n0 + c) * 512 + k0 + 32, 0, 3);
    v16h af[2];
#pragma unroll
    for (int r = 0; r < 2; ++r)
      af[r] = load_a_frag(O2 + (size_t)(m0 + r * 16 + c) * 512 + k0 + hi * 16);
    v16h bf[4];
#pragma unroll
    for (int t = 0; t < 4; ++t)
      bf[t] = *(const v16h*)(wh + (size_t)(n0 + t * 16 + c) * 512 + k0 + hi * 16);
#pragma unroll
    for (int r = 0; r < 2; ++r)
#pragma unroll
      for (int t = 0; t < 4; ++t) acc[r][t] = wmma_f16(af[r], bf[t], acc[r][t]);
  }
  float bcol[4];
#pragma unroll
  for (int t = 0; t < 4; ++t) bcol[t] = bias[n0 + t * 16 + c];
#pragma unroll
  for (int r = 0; r < 2; ++r)
#pragma unroll
    for (int t = 0; t < 4; ++t)
#pragma unroll
      for (int v = 0; v < 8; ++v) {
        const int m = m0 + r * 16 + hi * 8 + v;
        out[(size_t)m * 512 + n0 + t * 16 + c] = acc[r][t][v] + bcol[t];
      }
}

// ---------------------------------------------------------------------------
extern "C" void kernel_launch(void* const* d_in, const int* in_sizes, int n_in,
                              void* d_out, int out_size, void* d_ws, size_t ws_size,
                              hipStream_t stream) {
  const float* x         = (const float*)d_in[0];  // (4,1024,512)
  const float* pair_bias = (const float*)d_in[1];  // (4,1024,1024)
  const float* in_proj_w = (const float*)d_in[2];  // (1536,512)
  const float* in_proj_b = (const float*)d_in[3];  // (1536)
  const float* out_w     = (const float*)d_in[4];  // (512,512)
  const float* out_b     = (const float*)d_in[5];  // (512)
  const float* pair_w    = (const float*)d_in[6];  // (8,1)
  const float* pair_b    = (const float*)d_in[7];  // (8)

  const size_t NE  = (size_t)4 * 8 * 1024 * 64;    // 2,097,152
  const size_t NW1 = (size_t)1536 * 512;           //   786,432
  const size_t NW2 = (size_t)512 * 512;            //   262,144
  half_t* Q   = (half_t*)d_ws;
  half_t* Kt  = Q   + NE;
  half_t* Vt  = Kt  + NE;
  half_t* O2  = Vt  + NE;
  half_t* xh  = O2  + NE;
  half_t* w1h = xh  + NE;
  half_t* w2h = w1h + NW1;

  // 0) one-time f32 -> f16 conversions (x, in_proj_w, out_w)
  cvt_kernel<<<(int)(NE  / 2048), 256, 0, stream>>>(x,         xh,  (int)NE);
  cvt_kernel<<<(int)(NW1 / 2048), 256, 0, stream>>>(in_proj_w, w1h, (int)NW1);
  cvt_kernel<<<(int)(NW2 / 2048), 256, 0, stream>>>(out_w,     w2h, (int)NW2);
  // 1) QKV projection: grid (4096/128, 1536/64)
  qkv_kernel<<<dim3(32, 24), 128, 0, stream>>>(xh, w1h, in_proj_b, Q, Kt, Vt);
  // 2) flash attention: one block per (b*H+h, q-tile of 64)
  attn_kernel<<<dim3(32, 16), 128, 0, stream>>>(Q, Kt, Vt, pair_bias, pair_w, pair_b, O2);
  // 3) output projection: grid (4096/128, 512/64)
  outproj_kernel<<<dim3(32, 8), 128, 0, stream>>>(O2, w2h, out_b, (float*)d_out);
}
